// SeqAutoEncoder_28896539967612
// MI455X (gfx1250) — matrix operand order repaired and checked by reading, not compiled, not run on hardware
//
#include <hip/hip_runtime.h>
#include <hip/hip_bf16.h>

// ---------------------------------------------------------------------------
// Problem constants (from reference): B=64, T=2048, D=256, H=30, 4H=120, NOUT=5
// ---------------------------------------------------------------------------
#define BB    64
#define TT    2048
#define DD    256
#define HH    30
#define GG    120          // 4*H
#define NPAD  128          // padded gate width for WMMA tiles
#define NOUTV 5

typedef __attribute__((ext_vector_type(16))) _Float16 v16h;
typedef __attribute__((ext_vector_type(8)))  float    v8f;
typedef __attribute__((ext_vector_type(4)))  unsigned int v4u;

union FragAB { v16h v; v4u q[2]; };

__device__ __forceinline__ unsigned int pack2h(float a, float b) {
    union { _Float16 h[2]; unsigned int u; } x;
    x.h[0] = (_Float16)a; x.h[1] = (_Float16)b;
    return x.u;
}

__device__ __forceinline__ float sigm(float x) {
    return 1.0f / (1.0f + __expf(-x));
}

// ---------------------------------------------------------------------------
// Kernel 0: pack enc_W1 x-part (rows 0..255, cols 0..119, zero-pad to 128)
// into B-fragment order for v_wmma_f32_16x16x32_f16.
// B fragment layout (32x16, f16): lane l -> column n = (l&15); half = l>>4;
// dword v (0..7) holds K = ktile*32 + 16*half + 2v (+1).
// Flat index: ((kt*8 + nt)*32 + lane)*8 + v
// ---------------------------------------------------------------------------
__global__ void pack_w1_kernel(const float* __restrict__ W1,
                               unsigned int* __restrict__ wpack) {
    int idx = blockIdx.x * blockDim.x + threadIdx.x;   // 0 .. 16383
    if (idx >= 8 * 8 * 32 * 8) return;
    int v    = idx & 7;
    int lane = (idx >> 3) & 31;
    int nt   = (idx >> 8) & 7;
    int kt   = idx >> 11;
    int n    = nt * 16 + (lane & 15);
    int half = lane >> 4;
    int k0   = kt * 32 + half * 16 + 2 * v;
    float a = 0.f, b = 0.f;
    if (n < GG) {
        a = W1[(size_t)k0 * GG + n];
        b = W1[(size_t)(k0 + 1) * GG + n];
    }
    wpack[idx] = pack2h(a, b);
}

// ---------------------------------------------------------------------------
// Kernel 1: Zx[m, 0:128] = seq[m, 0:256] @ W1x[0:256, 0:128]   (m = b*T + t)
// Tiled 64 rows per workgroup, full 128 cols. 8 waves:
//   wave w: mt = w&3, nts = (w>>2)*4 .. +3  -> 4 accumulators of v8f.
// A staged in LDS as packed f16 pairs [64 rows][16 dwords], fragment = 2x b128.
// ---------------------------------------------------------------------------
__global__ void __launch_bounds__(256)
gemm_zx_kernel(const float* __restrict__ seq,
               const unsigned int* __restrict__ wpack,
               float* __restrict__ Zx) {
    __shared__ unsigned int Atile[64 * 16];

    const int tid  = threadIdx.x;
    const int lane = tid & 31;
    const int w    = tid >> 5;
    const int mt   = w & 3;
    const int nts  = (w >> 2) * 4;
    const int half = lane >> 4;
    const int m0   = blockIdx.x * 64;

    const v8f vzero = {0.f,0.f,0.f,0.f,0.f,0.f,0.f,0.f};
    v8f acc[4] = {vzero, vzero, vzero, vzero};

    const int arow = tid >> 2;        // 0..63
    const int acq  = tid & 3;         // 0..3 -> 8 K-columns each

    for (int kt = 0; kt < 8; ++kt) {
        // ---- stage A tile (64 x 32 f32 -> f16 pairs) ----
        {
            const float* src = seq + (size_t)(m0 + arow) * DD + kt * 32 + acq * 8;
            const float4* s4 = (const float4*)src;
            float4 f0 = s4[0];
            float4 f1 = s4[1];
            unsigned int* dst = &Atile[arow * 16 + acq * 4];
            dst[0] = pack2h(f0.x, f0.y);
            dst[1] = pack2h(f0.z, f0.w);
            dst[2] = pack2h(f1.x, f1.y);
            dst[3] = pack2h(f1.z, f1.w);
        }
        __syncthreads();

        // ---- A fragment: contiguous k = 16*half + s, same order as wpack ----
        FragAB afrag;
        {
            const int m = mt * 16 + (lane & 15);
            afrag.q[0] = *(const v4u*)&Atile[m * 16 + 8 * half];      // K 0-7 / 16-23
            afrag.q[1] = *(const v4u*)&Atile[m * 16 + 8 * half + 4];  // K 8-15 / 24-31
        }

        // ---- 4 B fragments + 4 WMMAs ----
        #pragma unroll
        for (int i = 0; i < 4; ++i) {
            const int nt = nts + i;
            FragAB bfrag;
            const v4u* bp = (const v4u*)(wpack + ((size_t)(kt * 8 + nt) * 32 + lane) * 8);
            bfrag.q[0] = bp[0];
            bfrag.q[1] = bp[1];
            acc[i] = __builtin_amdgcn_wmma_f32_16x16x32_f16(
                false, afrag.v, false, bfrag.v, (short)0, acc[i], false, false);
        }
        __syncthreads();
    }

    // ---- write D: VGPR r -> row (r + 8*half), col = lane&15 ----
    #pragma unroll
    for (int i = 0; i < 4; ++i) {
        const int n = (nts + i) * 16 + (lane & 15);
        #pragma unroll
        for (int r = 0; r < 8; ++r) {
            const int mrow = m0 + mt * 16 + half * 8 + r;
            Zx[(size_t)mrow * NPAD + n] = acc[i][r];
        }
    }
}

// ---------------------------------------------------------------------------
// Kernel 2: combined recurrences. Blocks 0..15: encoder (4 batch rows each,
// LSTM1 input projection comes from Zx). Block 16: the single decoder
// sequence (dec_in is all-zero => batch-invariant; only dec_W1 rows 1..30
// matter). One thread per (batch, hidden-unit) computes all 4 gates with
// register-cached h-state; 3 barriers/step; cell states live in registers.
// Zx reads for step t+1 are prefetched during step t to hide global latency.
// ---------------------------------------------------------------------------
__global__ void __launch_bounds__(128)
recurrence_kernel(const float* __restrict__ Zx,
                  const float* __restrict__ eW1, const float* __restrict__ eb1,
                  const float* __restrict__ eW2, const float* __restrict__ eb2,
                  const float* __restrict__ eWd, const float* __restrict__ ebd,
                  const float* __restrict__ dW1, const float* __restrict__ db1,
                  const float* __restrict__ dW2, const float* __restrict__ db2,
                  float* __restrict__ out_latent,   // d_out + B*T
                  float* __restrict__ dec_row)      // [T]
{
    __shared__ float Wh[HH * GG];        // layer-1 h-part weights
    __shared__ float W2s[2 * HH * GG];   // layer-2 weights
    __shared__ float bAs[GG], bBs[GG];
    __shared__ float Wds[HH * NOUTV];
    __shared__ float bds[NOUTV];
    __shared__ float h1s[4 * HH];
    __shared__ float h2s[4 * HH];

    const int  tid = threadIdx.x;
    const bool enc = (blockIdx.x < 16);
    const int  nb  = enc ? 4 : 1;          // batches handled by this block
    const int  b0  = blockIdx.x * 4;

    const float* srcWh = enc ? (eW1 + 256 * GG) : (dW1 + GG);  // skip x rows
    const float* srcW2 = enc ? eW2 : dW2;
    const float* srcb1 = enc ? eb1 : db1;
    const float* srcb2 = enc ? eb2 : db2;

    for (int i = tid; i < HH * GG; i += 128)     Wh[i]  = srcWh[i];
    for (int i = tid; i < 2 * HH * GG; i += 128) W2s[i] = srcW2[i];
    for (int i = tid; i < GG; i += 128)         { bAs[i] = srcb1[i]; bBs[i] = srcb2[i]; }
    if (enc) {
        for (int i = tid; i < HH * NOUTV; i += 128) Wds[i] = eWd[i];
        if (tid < NOUTV) bds[tid] = ebd[tid];
    }
    if (tid < 4 * HH) { h1s[tid] = 0.f; h2s[tid] = 0.f; }
    __syncthreads();

    const int  ub  = tid / HH;     // batch slot within block
    const int  uk  = tid % HH;     // hidden index
    const bool act = tid < nb * HH;
    float c1 = 0.f, c2 = 0.f;

    // prefetch Zx gate inputs for t = 0
    float zc0 = 0.f, zc1 = 0.f, zc2 = 0.f, zc3 = 0.f;
    if (enc && act) {
        const float* zp = Zx + (size_t)(b0 + ub) * TT * NPAD + uk;
        zc0 = zp[0]; zc1 = zp[HH]; zc2 = zp[2 * HH]; zc3 = zp[3 * HH];
    }

    for (int t = 0; t < TT; ++t) {
        // prefetch Zx for t+1 (hides global latency under this step's work)
        float zn0 = 0.f, zn1 = 0.f, zn2 = 0.f, zn3 = 0.f;
        if (enc && act && (t + 1 < TT)) {
            const float* zp = Zx + ((size_t)(b0 + ub) * TT + (t + 1)) * NPAD + uk;
            zn0 = zp[0]; zn1 = zp[HH]; zn2 = zp[2 * HH]; zn3 = zp[3 * HH];
        }

        float hr1[HH], hr2[HH];
        float ao = 0.f;
        if (act) {
            #pragma unroll
            for (int k = 0; k < HH; ++k) { hr1[k] = h1s[ub * HH + k]; hr2[k] = h2s[ub * HH + k]; }
            float ai = zc0 + bAs[uk];
            float aj = zc1 + bAs[HH + uk];
            float af = zc2 + bAs[2 * HH + uk];
            ao       = zc3 + bAs[3 * HH + uk];
            #pragma unroll
            for (int k = 0; k < HH; ++k) {
                const float h = hr1[k];
                ai += h * Wh[k * GG + uk];
                aj += h * Wh[k * GG + HH + uk];
                af += h * Wh[k * GG + 2 * HH + uk];
                ao += h * Wh[k * GG + 3 * HH + uk];
            }
            c1 = c1 * sigm(af + 1.0f) + sigm(ai) * tanhf(aj);
        }
        __syncthreads();                       // everyone done reading old h1s/h2s
        if (act) h1s[ub * HH + uk] = tanhf(c1) * sigm(ao);
        __syncthreads();                       // new h1 visible

        if (act) {
            #pragma unroll
            for (int k = 0; k < HH; ++k) hr1[k] = h1s[ub * HH + k];   // h1 (new)
            float bi = bBs[uk];
            float bj = bBs[HH + uk];
            float bf = bBs[2 * HH + uk];
            float bo = bBs[3 * HH + uk];
            #pragma unroll
            for (int k = 0; k < HH; ++k) {
                const float ha = hr1[k], hb = hr2[k];
                bi += ha * W2s[k * GG + uk]          + hb * W2s[(HH + k) * GG + uk];
                bj += ha * W2s[k * GG + HH + uk]     + hb * W2s[(HH + k) * GG + HH + uk];
                bf += ha * W2s[k * GG + 2 * HH + uk] + hb * W2s[(HH + k) * GG + 2 * HH + uk];
                bo += ha * W2s[k * GG + 3 * HH + uk] + hb * W2s[(HH + k) * GG + 3 * HH + uk];
            }
            c2 = c2 * sigm(bf + 1.0f) + sigm(bi) * tanhf(bj);
            const float nh2 = tanhf(c2) * sigm(bo);
            h2s[ub * HH + uk] = nh2;
            if (!enc && uk == HH - 1) dec_row[t] = nh2;   // d2[:, t, -1]
        }
        __syncthreads();                       // new h2 visible for next step

        zc0 = zn0; zc1 = zn1; zc2 = zn2; zc3 = zn3;
    }

    // latent = tanh(h2_final @ enc_Wd + enc_bd): 4 batches x 5 outputs
    if (enc && tid < 4 * NOUTV) {
        const int b = tid / NOUTV, n = tid % NOUTV;
        float acc = bds[n];
        #pragma unroll
        for (int k = 0; k < HH; ++k) acc += h2s[b * HH + k] * Wds[k * NOUTV + n];
        out_latent[(size_t)(b0 + b) * NOUTV + n] = tanhf(acc);
    }
}

// ---------------------------------------------------------------------------
// Kernel 3: broadcast decoder row to all 64 batch rows of d_out.
// ---------------------------------------------------------------------------
__global__ void __launch_bounds__(256)
dec_broadcast_kernel(const float* __restrict__ dec_row,
                     float* __restrict__ out) {
    const int idx = blockIdx.x * blockDim.x + threadIdx.x;   // b*T + t
    if (idx < BB * TT) out[idx] = dec_row[idx & (TT - 1)];
}

// ---------------------------------------------------------------------------
extern "C" void kernel_launch(void* const* d_in, const int* in_sizes, int n_in,
                              void* d_out, int out_size, void* d_ws, size_t ws_size,
                              hipStream_t stream) {
    const float* seq    = (const float*)d_in[0];
    const float* enc_W1 = (const float*)d_in[1];
    const float* enc_b1 = (const float*)d_in[2];
    const float* enc_W2 = (const float*)d_in[3];
    const float* enc_b2 = (const float*)d_in[4];
    const float* enc_Wd = (const float*)d_in[5];
    const float* enc_bd = (const float*)d_in[6];
    // d_in[7] dec_Wd, d_in[8] dec_bd: dead code in the reference (vec2 unused)
    const float* dec_W1 = (const float*)d_in[9];
    const float* dec_b1 = (const float*)d_in[10];
    const float* dec_W2 = (const float*)d_in[11];
    const float* dec_b2 = (const float*)d_in[12];

    float* out = (float*)d_out;                 // [B*T dec_out][B*NOUT latent]

    // workspace layout
    float*        Zx      = (float*)d_ws;                        // 131072*128 f32
    unsigned int* wpack   = (unsigned int*)(Zx + (size_t)BB * TT * NPAD); // 16384 u32
    float*        dec_row = (float*)(wpack + 8 * 8 * 32 * 8);    // 2048 f32

    // 0) pre-swizzle W1 x-part into WMMA B-fragment order
    pack_w1_kernel<<<64, 256, 0, stream>>>(enc_W1, wpack);

    // 1) big WMMA GEMM: Zx = seq @ W1x  (131072 x 256 x 128)
    gemm_zx_kernel<<<(BB * TT) / 64, 256, 0, stream>>>(seq, wpack, Zx);

    // 2) encoder (16 blocks x 4 batches) and decoder (block 16) recurrences,
    //    overlapped in a single launch; encoder also emits latent.
    recurrence_kernel<<<17, 128, 0, stream>>>(
        Zx, enc_W1, enc_b1, enc_W2, enc_b2, enc_Wd, enc_bd,
        dec_W1, dec_b1, dec_W2, dec_b2,
        out + (size_t)BB * TT, dec_row);

    // 3) broadcast decoder row to all batch rows
    dec_broadcast_kernel<<<(BB * TT) / 256, 256, 0, stream>>>(dec_row, out);
}
